// Attention_32263794328002
// MI455X (gfx1250) — matrix operand order repaired and checked by reading, not compiled, not run on hardware
//
#include <hip/hip_runtime.h>

// Problem constants (from reference): B=16, C=384, H=W=32 -> N=1024, HEADS=6, d=64
#define B_DIM  16
#define C_DIM  384
#define N_DIM  1024
#define HEADS  6
#define D_DIM  64

// ---------------------------------------------------------------------------
// CDNA5 WMMA types (wave32; 16x16x32 bf16 -> f32 accumulate)
// ---------------------------------------------------------------------------
typedef __attribute__((ext_vector_type(16))) __bf16 v16bf;
typedef __attribute__((ext_vector_type(8)))  float  v8f;

union FragU {                 // one WMMA A/B operand: 8 VGPRs = 16 bf16
  v16bf          bf;
  unsigned short us[16];
  uint4          q[2];
};

union Pack8 {                 // 8 bf16 packed for a single b128 store
  unsigned short us[8];
  uint4           q;
};

union Pack4 {                 // 4 bf16 packed for a single b64 store
  unsigned short us[4];
  uint2           q;
};

// f32 -> bf16 via the native conversion
__device__ __forceinline__ unsigned short f2bf(float f) {
  union { __bf16 h; unsigned short u; } c;
  c.h = (__bf16)f;
  return c.u;
}

// ---------------------------------------------------------------------------
// Kernel 0a: bulk f32 -> bf16 convert (W, layout preserved)
// ---------------------------------------------------------------------------
__global__ __launch_bounds__(256) void cvt_f32_bf16(
    const float* __restrict__ src, unsigned short* __restrict__ dst, int n4)
{
  int i = blockIdx.x * 256 + threadIdx.x;
  if (i >= n4) return;
  float4 f = ((const float4*)src)[i];
  Pack4 p;
  p.us[0] = f2bf(f.x); p.us[1] = f2bf(f.y);
  p.us[2] = f2bf(f.z); p.us[3] = f2bf(f.w);
  ((uint2*)dst)[i] = p.q;
}

// ---------------------------------------------------------------------------
// Kernel 0b: transpose-convert X: (b, c, n) f32 -> (b, n, c) bf16.
// Doing the transpose once here means the GEMM kernel's B fragments are
// contiguous 16B global loads: no LDS staging, no per-kstep barriers.
// ---------------------------------------------------------------------------
__global__ __launch_bounds__(256) void xpose_cvt(
    const float* __restrict__ x, unsigned short* __restrict__ xt)
{
  __shared__ float t[16][17];                 // +1 pad: bank-conflict-free
  const int n0 = blockIdx.x * 16;
  const int c0 = blockIdx.y * 16;
  const int b  = blockIdx.z;
  const int r  = threadIdx.x >> 4;
  const int q  = threadIdx.x & 15;
  t[r][q] = x[(size_t)(b * C_DIM + c0 + r) * N_DIM + n0 + q];   // coalesced in n
  __syncthreads();
  xt[(size_t)(b * N_DIM + n0 + r) * C_DIM + c0 + q] = f2bf(t[q][r]); // coalesced in c
}

// ---------------------------------------------------------------------------
// Kernel 1: QKV projection  qkv[o, b, n] = sum_c W[o,c] * Xt[b,n,c]
// One wave per 16(o) x 16(n) tile; fully unrolled K loop of 12 WMMAs; all
// operands are contiguous b128 loads with immediate offsets (no LDS).
//   Q stored (b,h,n,d) bf16 *pre-scaled by C^-0.5*; K stored (b,h,n,d);
//   V stored (b,h,d,n) so attention's V B-fragments are contiguous.
// ---------------------------------------------------------------------------
__global__ __launch_bounds__(128) void qkv_proj_wmma(
    const unsigned short* __restrict__ xt,
    const unsigned short* __restrict__ w_bf,
    unsigned short* __restrict__ q_ws, unsigned short* __restrict__ k_ws,
    unsigned short* __restrict__ v_ws)
{
  const int tid   = threadIdx.x;
  const int lane  = tid & 31;
  const int wave  = tid >> 5;
  const int hs    = lane >> 4;        // half-wave select
  const int col   = lane & 15;
  const int hsOff = hs * 8;

  const int ntile = blockIdx.x;             // 0..63
  const int otile = blockIdx.y * 4 + wave;  // 0..71
  const int b     = blockIdx.z;             // 0..15
  const int obase = otile * 16;
  const int nbase = ntile * 16;

  // hoisted base pointers: all k-loop accesses use immediate offsets
  const unsigned short* wr = w_bf + (size_t)(obase + col) * C_DIM + hsOff;
  const unsigned short* xr = xt + (size_t)(b * N_DIM + nbase + col) * C_DIM + hsOff;

  v8f acc = {0.f, 0.f, 0.f, 0.f, 0.f, 0.f, 0.f, 0.f};

#pragma unroll
  for (int ks = 0; ks < C_DIM; ks += 32) {
    // A fragment (W, 16x32): lane row = obase+col; K chunks hs*8 and 16+hs*8
    FragU a;
    a.q[0] = *(const uint4*)(wr + ks);
    a.q[1] = *(const uint4*)(wr + ks + 16);
    // B fragment (Xt, 32x16): lane col = n; K chunks hs*8 and 16+hs*8
    FragU bm;
    bm.q[0] = *(const uint4*)(xr + ks);
    bm.q[1] = *(const uint4*)(xr + ks + 16);
    acc = __builtin_amdgcn_wmma_f32_16x16x32_bf16(
        false, a.bf, false, bm.bf, (short)0, acc, false, false);
  }

  // C/D layout: vgpr r -> row obase + r + hsOff, col = nbase + col(lane).
  // Whole tile shares one (which, head): decode from wave-uniform obase,
  // pinned to SGPRs so branches are scalar.
  const float qscale = rsqrtf((float)C_DIM);
  const int which = __builtin_amdgcn_readfirstlane(obase / C_DIM);  // 0=Q 1=K 2=V
  const int rem   = __builtin_amdgcn_readfirstlane(obase % C_DIM);
  const int h     = rem / D_DIM;
  const int bh    = b * HEADS + h;
  const int dd0   = (rem % D_DIM) + hsOff;
  const int n     = nbase + col;

  if (which == 0) {
    Pack8 p;
#pragma unroll
    for (int r = 0; r < 8; ++r) p.us[r] = f2bf(acc[r] * qscale);
    *(uint4*)&q_ws[((size_t)bh * N_DIM + n) * D_DIM + dd0] = p.q;
  } else if (which == 1) {
    Pack8 p;
#pragma unroll
    for (int r = 0; r < 8; ++r) p.us[r] = f2bf(acc[r]);
    *(uint4*)&k_ws[((size_t)bh * N_DIM + n) * D_DIM + dd0] = p.q;
  } else {
    // V transposed (d-major)
#pragma unroll
    for (int r = 0; r < 8; ++r)
      v_ws[((size_t)bh * D_DIM + dd0 + r) * N_DIM + n] = f2bf(acc[r]);
  }
}

// ---------------------------------------------------------------------------
// Kernel 2: flash attention. One wave per (b,h, 16-query block).
// Per 32-key block: 4 WMMAs for S, row-max via shfl_xor, P restaged through
// wave-private LDS (D->A layout), 1 WMMA against an all-ones B fragment to
// get row sums (softmax denominator) on the matrix engine, 4 WMMAs for P.V.
// Mask streamed non-temporally; base pointers advanced by constants.
// ---------------------------------------------------------------------------
__global__ __launch_bounds__(128) void flash_attn_wmma(
    const unsigned short* __restrict__ q_ws,
    const unsigned short* __restrict__ k_ws,
    const unsigned short* __restrict__ v_ws,
    const unsigned char*  __restrict__ mask,
    float* __restrict__ out)
{
  __shared__ __align__(16) unsigned short plds[4][16 * 32]; // per-wave P tile

  const int tid   = threadIdx.x;
  const int lane  = tid & 31;
  const int wave  = tid >> 5;
  const int hs    = lane >> 4;
  const int col   = lane & 15;
  const int hsOff = hs * 8;

  const int bh    = blockIdx.x;                 // 0..95  (b*HEADS + h)
  const int qblk  = blockIdx.y * 4 + wave;      // 0..63
  const int qbase = qblk * 16;
  const int b     = bh / HEADS;
  const int h     = bh % HEADS;

  // Q A-fragments (row = qbase+col): dd 0..31 and 32..63, pre-scaled
  FragU qA0, qA1;
  {
    const unsigned short* qr = q_ws + ((size_t)bh * N_DIM + qbase + col) * D_DIM + hs * 16;
    qA0.q[0] = *(const uint4*)(qr + 0);
    qA0.q[1] = *(const uint4*)(qr + 8);
    qA1.q[0] = *(const uint4*)(qr + 32);
    qA1.q[1] = *(const uint4*)(qr + 40);
  }

  // constant all-ones B fragment: row-sum via the matrix engine
  FragU ones;
#pragma unroll
  for (int i = 0; i < 16; ++i) ones.us[i] = 0x3F80;  // bf16(1.0)

  const v8f vzero = {0.f, 0.f, 0.f, 0.f, 0.f, 0.f, 0.f, 0.f};
  v8f acc[4];
#pragma unroll
  for (int t = 0; t < 4; ++t) acc[t] = vzero;
  v8f lacc = vzero;                          // per-row softmax denominator

  float mrow[8];
#pragma unroll
  for (int r = 0; r < 8; ++r) mrow[r] = -3.0e38f;

  // hoisted base pointers, advanced by constants each key block
  const unsigned short* kp = k_ws + ((size_t)bh * N_DIM + col) * D_DIM + hs * 16;
  const unsigned short* vp = v_ws + ((size_t)bh * D_DIM + col) * N_DIM + hs * 16;
  const unsigned char*  mp = mask + ((size_t)bh * N_DIM + qbase + hsOff) * N_DIM + col;

  for (int j = 0; j < N_DIM; j += 32) {
    __builtin_prefetch(kp + 32 * D_DIM, 0, 1);  // next K block toward L2

    // ---- S = Q . K^T : two 16x16 tiles (key cols j..j+15, j+16..j+31)
    v8f s0 = vzero, s1 = vzero;
    {
      FragU kb;
      kb.q[0] = *(const uint4*)(kp + 0);    kb.q[1] = *(const uint4*)(kp + 8);
      s0 = __builtin_amdgcn_wmma_f32_16x16x32_bf16(false, qA0.bf, false, kb.bf, (short)0, s0, false, false);
      kb.q[0] = *(const uint4*)(kp + 32);   kb.q[1] = *(const uint4*)(kp + 40);
      s0 = __builtin_amdgcn_wmma_f32_16x16x32_bf16(false, qA1.bf, false, kb.bf, (short)0, s0, false, false);
      kb.q[0] = *(const uint4*)(kp + 1024); kb.q[1] = *(const uint4*)(kp + 1032);
      s1 = __builtin_amdgcn_wmma_f32_16x16x32_bf16(false, qA0.bf, false, kb.bf, (short)0, s1, false, false);
      kb.q[0] = *(const uint4*)(kp + 1056); kb.q[1] = *(const uint4*)(kp + 1064);
      s1 = __builtin_amdgcn_wmma_f32_16x16x32_bf16(false, qA1.bf, false, kb.bf, (short)0, s1, false, false);
    }

    // ---- dropout mask: True -> -inf (stream NT, read-once 100 MB)
#pragma unroll
    for (int r = 0; r < 8; ++r) {
      unsigned char m0 = __builtin_nontemporal_load(mp + r * N_DIM);
      unsigned char m1 = __builtin_nontemporal_load(mp + r * N_DIM + 16);
      if (m0) s0[r] = -1.0e30f;
      if (m1) s1[r] = -1.0e30f;
    }

    // ---- online softmax; rows live in 16-lane halves (row = r + hsOff)
#pragma unroll
    for (int r = 0; r < 8; ++r) {
      float tm = fmaxf(s0[r], s1[r]);
#pragma unroll
      for (int d = 8; d >= 1; d >>= 1) tm = fmaxf(tm, __shfl_xor(tm, d, 32));
      float mn = fmaxf(mrow[r], tm);
      float sc = __expf(mrow[r] - mn);
      float p0 = __expf(s0[r] - mn);
      float p1 = __expf(s1[r] - mn);
      mrow[r] = mn;
      acc[0][r] *= sc; acc[1][r] *= sc; acc[2][r] *= sc; acc[3][r] *= sc;
      lacc[r] *= sc;                          // rescale denominator too
      plds[wave][(r + hsOff) * 32 + col]      = f2bf(p0);
      plds[wave][(r + hsOff) * 32 + 16 + col] = f2bf(p1);
    }
    // wave-private LDS RAW: DS ops are in-order per wave; wait to be explicit
    asm volatile("s_wait_dscnt 0" ::: "memory");

    // ---- P A-fragment (row = col lane, K = key index within block)
    FragU pA;
    pA.q[0] = *(const uint4*)&plds[wave][col * 32 + hs * 16];
    pA.q[1] = *(const uint4*)&plds[wave][col * 32 + hs * 16 + 8];

    // ---- denominator: lacc += P . ones  (row sums, replicated per lane)
    lacc = __builtin_amdgcn_wmma_f32_16x16x32_bf16(
        false, pA.bf, false, ones.bf, (short)0, lacc, false, false);

    // ---- O += P . V  (V is d-major: contiguous 32B per B-fragment lane)
#pragma unroll
    for (int t = 0; t < 4; ++t) {
      FragU vB;
      vB.q[0] = *(const uint4*)(vp + t * 16 * N_DIM);
      vB.q[1] = *(const uint4*)(vp + t * 16 * N_DIM + 8);
      acc[t] = __builtin_amdgcn_wmma_f32_16x16x32_bf16(
          false, pA.bf, false, vB.bf, (short)0, acc[t], false, false);
    }

    kp += 32 * D_DIM;
    vp += 32;
    mp += 32;
  }

  // ---- normalize and store: out[b, h*64+dd, n], n contiguous per vgpr group
  float inv[8];
#pragma unroll
  for (int r = 0; r < 8; ++r) inv[r] = 1.0f / lacc[r];
#pragma unroll
  for (int t = 0; t < 4; ++t) {
    float4 lo, hi;
    lo.x = acc[t][0] * inv[0]; lo.y = acc[t][1] * inv[1];
    lo.z = acc[t][2] * inv[2]; lo.w = acc[t][3] * inv[3];
    hi.x = acc[t][4] * inv[4]; hi.y = acc[t][5] * inv[5];
    hi.z = acc[t][6] * inv[6]; hi.w = acc[t][7] * inv[7];
    float* op = out + (size_t)(b * C_DIM + h * D_DIM + t * 16 + col) * N_DIM
                    + qbase + hsOff;
    *(float4*)op       = lo;
    *(float4*)(op + 4) = hi;
  }
}

// ---------------------------------------------------------------------------
// Launch: inputs are {x (f32), w_qkv (f32), drop_mask (bool/u8)}; out f32.
// Workspace: Q,K,V bf16 (37.7 MB) + W,Xt bf16 (13.5 MB) = ~51 MB.
// ---------------------------------------------------------------------------
extern "C" void kernel_launch(void* const* d_in, const int* in_sizes, int n_in,
                              void* d_out, int out_size, void* d_ws, size_t ws_size,
                              hipStream_t stream) {
  (void)in_sizes; (void)n_in; (void)out_size; (void)ws_size;

  const float*         x    = (const float*)d_in[0];
  const float*         w    = (const float*)d_in[1];
  const unsigned char* mask = (const unsigned char*)d_in[2];
  float*               out  = (float*)d_out;

  const size_t qkv_elems = (size_t)B_DIM * HEADS * N_DIM * D_DIM;  // 6.29M
  const size_t w_elems   = (size_t)3 * C_DIM * C_DIM;              // 442368

  unsigned short* q_ws = (unsigned short*)d_ws;
  unsigned short* k_ws = q_ws + qkv_elems;
  unsigned short* v_ws = k_ws + qkv_elems;
  unsigned short* w_bf = v_ws + qkv_elems;
  unsigned short* x_bf = w_bf + w_elems;     // transposed: (b, n, c)

  // Kernel 0: one-time conversions (W linear; X transpose-convert)
  {
    int w4 = (int)(w_elems / 4);
    cvt_f32_bf16<<<dim3((w4 + 255) / 256), 256, 0, stream>>>(w, w_bf, w4);
    dim3 gX(N_DIM / 16, C_DIM / 16, B_DIM);
    xpose_cvt<<<gX, 256, 0, stream>>>(x, x_bf);
  }

  // Kernel 1: grid = (ntile=64, otile-group=72/4=18, b=16), 4 waves/block
  dim3 gA(N_DIM / 16, (3 * C_DIM / 16) / 4, B_DIM);
  qkv_proj_wmma<<<gA, 128, 0, stream>>>(x_bf, w_bf, q_ws, k_ws, v_ws);

  // Kernel 2: grid = (b*h=96, qblock-group=64/4=16), 4 waves/block
  dim3 gB(B_DIM * HEADS, (N_DIM / 16) / 4, 1);
  flash_attn_wmma<<<gB, 128, 0, stream>>>(q_ws, k_ws, v_ws, mask, out);
}
